// att_7215545057282
// MI455X (gfx1250) — matrix operand order repaired and checked by reading, not compiled, not run on hardware
//
#include <hip/hip_runtime.h>
#include <hip/hip_bf16.h>
#include <math.h>

// ---------------------------------------------------------------------------
// Persistent bf16-WMMA LSTM/attention recurrence for MI455X (gfx1250).
//
//   T=2048 steps, B=64, H=256.  Per step:
//     pre   = tanh(x_t W_b^T + [h,c] W_e^T + b_b + b_e)        [64,256]
//     j     = pre . v_b ; s = softmax_over_batch(j)            [64]
//     gates = (s*x_t) W_ih^T + h W_hh^T + b_ih + b_hh          [64,1024]
//     LSTM pointwise -> h,c ; out[t] = h
//
//   16 persistent workgroups (one 16-col slice of H each), 8 waves of 32.
//   bf16 WMMA (v_wmma_f32_16x16x32_bf16), fp32 accumulation.  Both A and B
//   operands are pre-swizzled into WMMA fragment order: B once at launch
//   (global, L2-resident), A per step into LDS at bf16.  Softmax scaling is
//   applied with v_pk_mul_bf16 directly on the packed fragment.
// ---------------------------------------------------------------------------

typedef __attribute__((ext_vector_type(16))) __bf16 v16bf;
typedef __attribute__((ext_vector_type(8)))  float  v8f;

#define TT   2048
#define BB   64
#define HH   256
#define NWG  16
#define NTHR 256

// ---- WMMA wrapper ----------------------------------------------------------
__device__ __forceinline__ v8f wmma_bf16(v16bf a, v16bf b, v8f c) {
  return __builtin_amdgcn_wmma_f32_16x16x32_bf16(
      false, a, false, b, (short)0, c, false, false);
}

// ---- fragment-order helpers ------------------------------------------------
// A-tile (16x32 bf16) fragment slot for (mt,kt): 32 lanes x 16 elems = 1 KB.
// Lane l holds row m=(l&15); elems 0..7 = K[k0..k0+7], elems 8..15 =
// K[k0+16..k0+23] with k0 = kt*32 + 8*(l>>4)  (ISA 7.12.2, 16-bit A 16x32).
__device__ __forceinline__ const __bf16* frag_slot(const __bf16* base, int mt,
                                                   int kt, int lane) {
  return base + (((mt * 8 + kt) * 32) + lane) * 16;
}

__device__ __forceinline__ v16bf load_a_frag(const __bf16* base, int mt, int kt,
                                             int lane) {
  return *(const v16bf*)frag_slot(base, mt, kt, lane);
}

// Build one packed A fragment slot from fp32 global row-major source.
__device__ __forceinline__ void build_frag(const float* __restrict__ g,
                                           __bf16* dst, int m, int k0) {
  const float* r = g + m * HH + k0;
  v16bf v;
#pragma unroll
  for (int e = 0; e < 8; ++e) v[e] = (__bf16)r[e];
#pragma unroll
  for (int e = 0; e < 8; ++e) v[8 + e] = (__bf16)r[16 + e];
  *(v16bf*)dst = v;
}

// Multiply a packed bf16 fragment by a row-broadcast scale (v_pk_mul_bf16).
__device__ __forceinline__ v16bf scale_frag(v16bf a, unsigned sdup) {
  union { v16bf v; unsigned u[8]; } in, out;
  in.v = a;
#pragma unroll
  for (int i = 0; i < 8; ++i)
    asm("v_pk_mul_bf16 %0, %1, %2" : "=v"(out.u[i]) : "v"(in.u[i]), "v"(sdup));
  return out.v;
}

// B-fragment: pre-swizzled in global memory, 32B/lane coalesced b128 loads.
__device__ __forceinline__ v16bf frag_b(const __bf16* packed, int KT, int nt,
                                        int kt, int lane) {
  return *(const v16bf*)(packed + (((size_t)nt * KT + kt) * 32 + lane) * 16);
}

__device__ __forceinline__ float sigmoidf_(float x) {
  return 1.0f / (1.0f + __expf(-x));
}

// ---- grid-wide sense barrier (16 resident WGs) -----------------------------
__device__ __forceinline__ void grid_barrier(unsigned* cnt, unsigned* gen) {
  __syncthreads();
  if (threadIdx.x == 0) {
    __threadfence();
    unsigned g = *(volatile unsigned*)gen;
    if (atomicAdd(cnt, 1u) == NWG - 1u) {
      *(volatile unsigned*)cnt = 0u;
      __threadfence();
      atomicAdd(gen, 1u);
    } else {
      while (*(volatile unsigned*)gen == g) __builtin_amdgcn_s_sleep(1);
      __threadfence();
    }
  }
  __syncthreads();
}

// ---------------------------------------------------------------------------
// Weight pack: W[N][K] fp32 (PyTorch [out,in]) -> bf16 B-fragments.
// ---------------------------------------------------------------------------
__global__ void __launch_bounds__(256) pack_b_kernel(const float* __restrict__ W,
                                                     __bf16* __restrict__ out,
                                                     int N, int K) {
  int KT    = K / 32;
  int total = (N / 16) * KT * 32;
  int idx   = blockIdx.x * blockDim.x + threadIdx.x;
  if (idx >= total) return;
  int lane = idx & 31;
  int tile = idx >> 5;
  int kt = tile % KT, nt = tile / KT;
  int n     = nt * 16 + (lane & 15);
  int kbase = kt * 32 + 8 * (lane >> 4);
  __bf16* dst = out + (size_t)idx * 16;
  const float* src = W + (size_t)n * K + kbase;
#pragma unroll
  for (int e = 0; e < 16; ++e) dst[e] = (__bf16)src[e < 8 ? e : e + 8];
}

__global__ void __launch_bounds__(256) init_state_kernel(float* hg, float* cg,
                                                         float* jbuf, unsigned* bar) {
  int i = blockIdx.x * blockDim.x + threadIdx.x;
  for (int k = i; k < BB * HH; k += gridDim.x * blockDim.x) { hg[k] = 0.f; cg[k] = 0.f; }
  if (i < 2 * BB) jbuf[i] = 0.f;
  if (i < 2) bar[i] = 0u;
}

// ---------------------------------------------------------------------------
// Persistent recurrence kernel.
// ---------------------------------------------------------------------------
__global__ void __launch_bounds__(NTHR, 1) rnn_persistent(
    const float* __restrict__ x,
    const float* __restrict__ b_b,  const float* __restrict__ v_b,
    const float* __restrict__ b_e,  const float* __restrict__ b_ih,
    const float* __restrict__ b_hh,
    const __bf16* __restrict__ pWb, const __bf16* __restrict__ pWe,
    const __bf16* __restrict__ pWih, const __bf16* __restrict__ pWhh,
    float* hg, float* cg, float* jbuf, unsigned* bar, float* out) {
  extern __shared__ __align__(16) char smem[];
  __bf16* s_xp   = (__bf16*)(smem);            // 32 tiles x 1KB = 32 KB
  __bf16* s_hp   = (__bf16*)(smem + 32768);    // 32 KB
  __bf16* s_cp   = (__bf16*)(smem + 65536);    // 32 KB
  float*  s_pre  = (float*) (smem + 98304);    // [64][16]  4 KB
  float*  s_gates= (float*) (smem + 102400);   // [4][64][16] 16 KB
  float*  s_js   = (float*) (smem + 118784);   // [64]
  float*  s_ss   = (float*) (smem + 119040);   // [64]
  float*  s_red  = (float*) (smem + 119296);   // max, 1/sum

  const int wg   = blockIdx.x;                 // owns H cols [wg*16, wg*16+16)
  const int tid  = threadIdx.x;
  const int lane = tid & 31;
  const int wid  = tid >> 5;
  const int n0   = wg * 16;
  unsigned* cnt = bar;
  unsigned* gen = bar + 1;

  const float bias_pre = b_b[n0 + (lane & 15)] + b_e[n0 + (lane & 15)];

  for (int t = 0; t < TT; ++t) {
    // Zero the j slot used at step t+1 (its last readers finished before the
    // final barrier of step t-1).
    if (wg == 0 && tid < BB) jbuf[((t + 1) & 1) * BB + tid] = 0.f;

    // ---- stage x_t, h, c as bf16 A-fragments in LDS (96 KB total) ----
    {
      const float* xt_g = x + (size_t)t * BB * HH;
      for (int s = tid; s < 32 * 32; s += NTHR) {    // (tile, lane) slots
        int tile = s >> 5, ln = s & 31;
        int mt = tile >> 3, kt = tile & 7;
        int m  = mt * 16 + (ln & 15);
        int k0 = kt * 32 + 8 * (ln >> 4);
        build_frag(xt_g, s_xp + (size_t)s * 16, m, k0);
        build_frag(hg,   s_hp + (size_t)s * 16, m, k0);
        build_frag(cg,   s_cp + (size_t)s * 16, m, k0);
      }
    }
    __syncthreads();

    // ---- phase A: pre slice [64 x 16], K = 256 (x@Wb) + 512 (hc@We) ----
    if (wid < 4) {
      const int mt = wid;
      v8f acc = {};
      for (int kt = 0; kt < 8; ++kt) {
        v16bf ax = load_a_frag(s_xp, mt, kt, lane);
        acc = wmma_bf16(ax, frag_b(pWb, 8, wg, kt, lane), acc);
        v16bf ah = load_a_frag(s_hp, mt, kt, lane);
        acc = wmma_bf16(ah, frag_b(pWe, 16, wg, kt, lane), acc);
        v16bf ac = load_a_frag(s_cp, mt, kt, lane);
        acc = wmma_bf16(ac, frag_b(pWe, 16, wg, kt + 8, lane), acc);
      }
      const int n = lane & 15, half = lane >> 4;
#pragma unroll
      for (int r = 0; r < 8; ++r) {
        int m = mt * 16 + r + 8 * half;              // C/D layout (ISA 7.12.2)
        s_pre[m * 16 + n] = tanhf(acc[r] + bias_pre);
      }
    }
    __syncthreads();

    // partial j = pre . v_b over this WG's 16 columns -> global accumulate
    if (tid < BB) {
      float pj = 0.f;
#pragma unroll
      for (int n = 0; n < 16; ++n) pj += s_pre[tid * 16 + n] * v_b[n0 + n];
      atomicAdd(&jbuf[(t & 1) * BB + tid], pj);      // global_atomic_add_f32
    }

    // prefetch next timestep's x while waiting (global_prefetch_b8)
    if (t + 1 < TT) {
      const char* nx = (const char*)(x + (size_t)(t + 1) * BB * HH);
      __builtin_prefetch(nx + tid * 256, 0, 1);
    }

    grid_barrier(cnt, gen);                          // sync #1: j complete

    // ---- softmax over batch (64 values, redundantly per WG) ----
    if (tid < BB) s_js[tid] = jbuf[(t & 1) * BB + tid];
    __syncthreads();
    if (tid == 0) {
      float mx = -1e30f;
      for (int b = 0; b < BB; ++b) mx = fmaxf(mx, s_js[b]);
      float sum = 0.f;
      for (int b = 0; b < BB; ++b) sum += __expf(s_js[b] - mx);
      s_red[0] = mx; s_red[1] = 1.f / sum;
    }
    __syncthreads();
    if (tid < BB) s_ss[tid] = __expf(s_js[tid] - s_red[0]) * s_red[1];
    __syncthreads();

    // ---- phase B: gates [64 x 64 slice] = (s*x) Wih^T + h Whh^T ----
    // Wave handles gates g0 and g0+2 with the SAME mtile: one A-fragment pair
    // feeds 4 WMMAs per kt.
    {
      const int mt = wid & 3;
      const int g0 = wid >> 2;                       // 0..1
      const int nt0 = g0 * 16 + wg;
      const int nt1 = (g0 + 2) * 16 + wg;
      // row-broadcast softmax scale, duplicated into both bf16 halves
      __bf16 sb = (__bf16)s_ss[mt * 16 + (lane & 15)];
      unsigned short us; __builtin_memcpy(&us, &sb, 2);
      const unsigned sdup = (unsigned)us | ((unsigned)us << 16);

      v8f acc0 = {}, acc1 = {};
      for (int kt = 0; kt < 8; ++kt) {
        v16bf aw = scale_frag(load_a_frag(s_xp, mt, kt, lane), sdup);
        v16bf ah = load_a_frag(s_hp, mt, kt, lane);
        acc0 = wmma_bf16(aw, frag_b(pWih, 8, nt0, kt, lane), acc0);
        acc0 = wmma_bf16(ah, frag_b(pWhh, 8, nt0, kt, lane), acc0);
        acc1 = wmma_bf16(aw, frag_b(pWih, 8, nt1, kt, lane), acc1);
        acc1 = wmma_bf16(ah, frag_b(pWhh, 8, nt1, kt, lane), acc1);
      }
      const int n = lane & 15, half = lane >> 4;
      {
        int gn = g0 * 256 + n0 + n;
        float bias = b_ih[gn] + b_hh[gn];
#pragma unroll
        for (int r = 0; r < 8; ++r)
          s_gates[(g0 * 64 + mt * 16 + r + 8 * half) * 16 + n] = acc0[r] + bias;
      }
      {
        int g1 = g0 + 2;
        int gn = g1 * 256 + n0 + n;
        float bias = b_ih[gn] + b_hh[gn];
#pragma unroll
        for (int r = 0; r < 8; ++r)
          s_gates[(g1 * 64 + mt * 16 + r + 8 * half) * 16 + n] = acc1[r] + bias;
      }
    }
    __syncthreads();

    // ---- LSTM pointwise update for this WG's [64 x 16] slice ----
    for (int e = tid; e < BB * 16; e += NTHR) {
      int b = e >> 4, n = e & 15;
      float ig = sigmoidf_(s_gates[(0 * 64 + b) * 16 + n]);
      float fg = sigmoidf_(s_gates[(1 * 64 + b) * 16 + n]);
      float gg = tanhf   (s_gates[(2 * 64 + b) * 16 + n]);
      float og = sigmoidf_(s_gates[(3 * 64 + b) * 16 + n]);
      int   gi   = b * HH + n0 + n;
      float cnew = fg * cg[gi] + ig * gg;            // fp32 c from L2-hot global
      float hnew = og * tanhf(cnew);
      cg[gi] = cnew;
      hg[gi] = hnew;
      out[(size_t)t * BB * HH + gi] = hnew;
    }

    grid_barrier(cnt, gen);                          // sync #2: h,c published
  }
}

// ---------------------------------------------------------------------------
// Host entry.
// ---------------------------------------------------------------------------
extern "C" void kernel_launch(void* const* d_in, const int* in_sizes, int n_in,
                              void* d_out, int out_size, void* d_ws, size_t ws_size,
                              hipStream_t stream) {
  const float* x    = (const float*)d_in[0];
  const float* W_b  = (const float*)d_in[1];
  const float* b_b  = (const float*)d_in[2];
  const float* v_b  = (const float*)d_in[3];
  const float* W_e  = (const float*)d_in[4];
  const float* b_e  = (const float*)d_in[5];
  const float* W_ih = (const float*)d_in[6];
  const float* W_hh = (const float*)d_in[7];
  const float* b_ih = (const float*)d_in[8];
  const float* b_hh = (const float*)d_in[9];
  float* out = (float*)d_out;

  char* ws = (char*)d_ws;
  __bf16*  pWb  = (__bf16*)(ws + 0);         //  256*256*2 = 128 KB
  __bf16*  pWe  = (__bf16*)(ws + 131072);    //  256*512*2 = 256 KB
  __bf16*  pWih = (__bf16*)(ws + 393216);    // 1024*256*2 = 512 KB
  __bf16*  pWhh = (__bf16*)(ws + 917504);    // 1024*256*2 = 512 KB
  float*   hg   = (float*) (ws + 1441792);   // 64 KB
  float*   cg   = (float*) (ws + 1507328);   // 64 KB
  float*   jbuf = (float*) (ws + 1572864);   // 2*64*4
  unsigned* bar = (unsigned*)(ws + 1573376); // count, generation

  pack_b_kernel<<<(256/16*256/32*32 + 255) / 256, 256, 0, stream>>>(W_b,  pWb,  256, 256);
  pack_b_kernel<<<(256/16*512/32*32 + 255) / 256, 256, 0, stream>>>(W_e,  pWe,  256, 512);
  pack_b_kernel<<<(1024/16*256/32*32 + 255) / 256, 256, 0, stream>>>(W_ih, pWih, 1024, 256);
  pack_b_kernel<<<(1024/16*256/32*32 + 255) / 256, 256, 0, stream>>>(W_hh, pWhh, 1024, 256);

  init_state_kernel<<<64, 256, 0, stream>>>(hg, cg, jbuf, bar);

  const size_t smem_bytes = 119360;          // ~117 KB dynamic LDS
  rnn_persistent<<<NWG, NTHR, smem_bytes, stream>>>(
      x, b_b, v_b, b_e, b_ih, b_hh, pWb, pWe, pWih, pWhh,
      hg, cg, jbuf, bar, out);
}
